// MemTransformerLM_57724360458870
// MI455X (gfx1250) — compile-verified, matrix operand order checked by reading, not run on hardware
//
#include <hip/hip_runtime.h>
#include <hip/hip_bf16.h>
#include <stddef.h>

// ---------------------------------------------------------------------------
// Problem constants (reference: SEQ=1536, BATCH=2, D_MODEL=1024, 16 heads x 64,
// DPFP n_roll=3 -> feature dim F = 2*64*3 = 384)
// ---------------------------------------------------------------------------
#define SEQ     1536
#define BATCH   2
#define DMODEL  1024
#define NHEAD   16
#define DHEAD   64
#define FDIM    384
#define MROWS   (SEQ * BATCH)      /* 3072 rows of the token matrix            */
#define QKVCOLS (3 * DMODEL)       /* 3072: [Q | K | V] concatenated columns   */
#define SCALEF  (0.125f)           /* 1/sqrt(64)                               */
#define EPSD    (1e-5f)
#define EPSLN   (1e-5f)

typedef __attribute__((ext_vector_type(16))) __bf16       v16bf;
typedef __attribute__((ext_vector_type(8)))  float        v8f;
typedef __attribute__((ext_vector_type(4)))  unsigned int u32x4;

// TDM descriptor vector types (probe-verified builtin signatures)
typedef __attribute__((ext_vector_type(4))) unsigned int tdm_u32x4;
typedef __attribute__((ext_vector_type(8))) int          tdm_i32x8;
typedef __attribute__((ext_vector_type(4))) int          tdm_i32x4;

// ---------------------------------------------------------------------------
// bf16 helpers (round-to-nearest-even)
// ---------------------------------------------------------------------------
__device__ __forceinline__ unsigned short f2bf(float x) {
  union { float f; unsigned int u; } c; c.f = x;
  unsigned int u = c.u;
  unsigned int r = u + 0x7FFFu + ((u >> 16) & 1u);
  if ((u & 0x7F800000u) == 0x7F800000u) r = u;   // inf/nan passthrough
  return (unsigned short)(r >> 16);
}
__device__ __forceinline__ float bf2f(unsigned short h) {
  union { unsigned int u; float f; } c; c.u = ((unsigned int)h) << 16;
  return c.f;
}

// ---------------------------------------------------------------------------
// WMMA fragment loads (ISA 7.12.2 16-bit layouts).
// A fragment (16x32, row m = lane%16): elems 0..7 <-> K = hl*8 + 0..7,
// elems 8..15 <-> K = 16 + hl*8 + 0..7 (hl = lane/16).  B fragment (32x16,
// col n = lane%16) uses the same K packing, i.e. a row of W for C = X @ W^T.
// ---------------------------------------------------------------------------
__device__ __forceinline__ v16bf load_frag(const unsigned short* p, int hl) {
  union { v16bf v; u32x4 q[2]; } f;
  f.q[0] = *(const u32x4*)(p + hl * 8);
  f.q[1] = *(const u32x4*)(p + 16 + hl * 8);
  return f.v;
}

__device__ __forceinline__ v8f wmma_bf16(v16bf a, v16bf b, v8f c) {
  return __builtin_amdgcn_wmma_f32_16x16x32_bf16(
      /*neg_a=*/false, a, /*neg_b=*/false, b,
      /*c_mod=*/(short)0, c, /*reuse_a=*/false, /*reuse_b=*/false);
}

// ---------------------------------------------------------------------------
// Kernel 1: cast inputs to bf16.  Wq (1024x1024) and Wkv (2048x1024) are
// concatenated row-wise into wqkv (3072x1024) so QKV is a single GEMM.
// ---------------------------------------------------------------------------
__global__ void cast_in_kernel(const float* __restrict__ h,
                               const float* __restrict__ Wq,
                               const float* __restrict__ Wkv,
                               const float* __restrict__ Wo,
                               unsigned short* __restrict__ hbf,
                               unsigned short* __restrict__ wqkv,
                               unsigned short* __restrict__ wo) {
  const int N1 = MROWS * DMODEL;            // h
  const int N2 = N1 + DMODEL * DMODEL;      // Wq
  const int N3 = N2 + 2 * DMODEL * DMODEL;  // Wkv
  const int N4 = N3 + DMODEL * DMODEL;      // Wo
  for (int i = blockIdx.x * blockDim.x + threadIdx.x; i < N4;
       i += gridDim.x * blockDim.x) {
    if (i < N1)       hbf[i]            = f2bf(h[i]);
    else if (i < N2)  wqkv[i - N1]      = f2bf(Wq[i - N1]);
    else if (i < N3)  wqkv[i - N1]      = f2bf(Wkv[i - N2]);
    else              wo[i - N3]        = f2bf(Wo[i - N3]);
  }
}

// ---------------------------------------------------------------------------
// Kernel 2/7: bf16 WMMA GEMM, C[M,N] = A[M,K] @ B[N,K]^T  (f32 accumulate).
// One wave computes a 16x64 output tile (A fragment reused by 4 B fragments).
// ---------------------------------------------------------------------------
__global__ __launch_bounds__(256) void gemm_bf16_nt(
    const unsigned short* __restrict__ A, const unsigned short* __restrict__ B,
    float* __restrict__ C, int M, int N, int K) {
  int wave  = blockIdx.x * (blockDim.x >> 5) + (threadIdx.x >> 5);
  int lane  = threadIdx.x & 31;
  int hl    = lane >> 4;
  int ln    = lane & 15;
  int ntile = N >> 6;                         // 64-wide column tiles
  int total = (M >> 4) * ntile;
  if (wave >= total) return;
  int tm = wave / ntile, tn = wave % ntile;

  const unsigned short* arow = A + (size_t)(tm * 16 + ln) * K;
  const unsigned short* br0  = B + (size_t)(tn * 64 + ln) * K;
  const unsigned short* br1  = br0 + (size_t)16 * K;
  const unsigned short* br2  = br0 + (size_t)32 * K;
  const unsigned short* br3  = br0 + (size_t)48 * K;

  v8f a0 = {}, a1 = {}, a2 = {}, a3 = {};
  for (int k0 = 0; k0 < K; k0 += 32) {
    __builtin_prefetch(arow + k0 + 128, 0, 0);      // global_prefetch_b8
    v16bf fa = load_frag(arow + k0, hl);
    v16bf b0 = load_frag(br0 + k0, hl);
    v16bf b1 = load_frag(br1 + k0, hl);
    v16bf b2 = load_frag(br2 + k0, hl);
    v16bf b3 = load_frag(br3 + k0, hl);
    a0 = wmma_bf16(fa, b0, a0);
    a1 = wmma_bf16(fa, b1, a1);
    a2 = wmma_bf16(fa, b2, a2);
    a3 = wmma_bf16(fa, b3, a3);
  }
  // C/D layout: n = lane&15, m = v + 8*(lane>>4)
  for (int v = 0; v < 8; ++v) {
    int m = v + 8 * hl;
    float* crow = C + (size_t)(tm * 16 + m) * N + tn * 64;
    crow[ln]      = a0[v];
    crow[16 + ln] = a1[v];
    crow[32 + ln] = a2[v];
    crow[48 + ln] = a3[v];
  }
}

// ---------------------------------------------------------------------------
// Kernel 3: DPFP feature map.  For one (token row r, head): build
// x[128] = [relu(q), relu(-q)], then f[(i-1)*128 + p] = x[p] * x[(p-i)&127],
// i = 1..3.  SCALE folded into the Q features.  Output layout is per-(b,head):
// dst[((b*16+head)*SEQ + s)*FDIM + f]  (bf16) so attention GEMMs are contiguous.
// ---------------------------------------------------------------------------
__global__ __launch_bounds__(128) void dpfp_kernel(
    const float* __restrict__ qkv, int colOff,
    unsigned short* __restrict__ dst, float scale) {
  __shared__ float xs[128];
  int rn = blockIdx.x;
  int r = rn >> 4, head = rn & 15;
  int t = threadIdx.x;
  float v = qkv[(size_t)r * QKVCOLS + colOff + head * DHEAD + (t & 63)];
  xs[t] = (t < 64) ? fmaxf(v, 0.f) : fmaxf(-v, 0.f);
  __syncthreads();
  int b = r & 1, s = r >> 1;
  size_t base = ((size_t)(b * NHEAD + head) * SEQ + s) * FDIM;
  float xt = xs[t];
  for (int i = 1; i <= 3; ++i) {
    float val = xt * xs[(t - i) & 127] * scale;
    dst[base + (i - 1) * 128 + t] = f2bf(val);
  }
}

// ---------------------------------------------------------------------------
// Kernel 4: V -> bf16 in per-(b,head) layout: vbn[((b*16+h)*SEQ + s)*64 + d]
// ---------------------------------------------------------------------------
__global__ void vcast_kernel(const float* __restrict__ qkv,
                             unsigned short* __restrict__ vbn) {
  const int total = MROWS * DMODEL;
  for (int i = blockIdx.x * blockDim.x + threadIdx.x; i < total;
       i += gridDim.x * blockDim.x) {
    int r = i >> 10, c = i & 1023;
    int head = c >> 6, d = c & 63;
    int b = r & 1, s = r >> 1;
    vbn[((size_t)(b * NHEAD + head) * SEQ + s) * DHEAD + d] =
        f2bf(qkv[(size_t)r * QKVCOLS + 2 * DMODEL + c]);
  }
}

// ---------------------------------------------------------------------------
// Kernel 5: streaming causal linear attention.
// Block = 4 waves; handles one (b,head) and 64 query rows (4 waves x 16).
// Q fragments are register-resident across the key loop (12 x v16bf = 96
// VGPRs).  The 24 KB Kf tile (32 contiguous rows) is staged into LDS by the
// Tensor Data Mover when available (1D D#, data_size=2B), else by cooperative
// b128 loads.  Per key tile of 32: 24 score WMMAs (K=384), causal mask in the
// accumulator, bf16 round-trip through LDS back into A-layout, 4 WMMAs for
// S@V (V transposed in LDS so B-frags are contiguous rows).
// ---------------------------------------------------------------------------
__global__ __launch_bounds__(128) void attn_kernel(
    const unsigned short* __restrict__ Qf, const unsigned short* __restrict__ Kf,
    const unsigned short* __restrict__ Vbn, float* __restrict__ av) {
  __shared__ __align__(16) unsigned short lKf[32 * FDIM];   // 24 KB
  __shared__ __align__(16) unsigned short lVt[DHEAD * 32];  //  4 KB (transposed)
  __shared__ __align__(16) unsigned short lSc[4][16 * 32];  //  4 KB
  __shared__ float lDen[4][16];

  const int QG = 24;                    // 96 qtiles / 4 waves
  int bn = blockIdx.x / QG;
  int qg = blockIdx.x % QG;
  int w    = threadIdx.x >> 5;
  int lane = threadIdx.x & 31;
  int hl   = lane >> 4, ln = lane & 15;
  int qbase = qg * 64 + w * 16;
  int b = bn >> 4, head = bn & 15;

  const unsigned short* qrow = Qf + ((size_t)bn * SEQ + qbase + ln) * FDIM;
  const unsigned short* kbp  = Kf + (size_t)bn * SEQ * FDIM;
  const unsigned short* vbp  = Vbn + (size_t)bn * SEQ * DHEAD;

  // register-resident Q fragments (loop invariant)
  v16bf qfr[FDIM / 32];
  for (int kk = 0; kk < FDIM / 32; ++kk) qfr[kk] = load_frag(qrow + kk * 32, hl);

  if (lane < 16) lDen[w][lane] = 0.f;
  v8f n0 = {}, n1 = {}, n2 = {}, n3 = {};

  int nkt = 2 * (qg + 1);               // key tiles of 32 covering j <= qg*64+63
  for (int jt = 0; jt < nkt; ++jt) {
    int kb = jt * 32;
    __syncthreads();
    // --- stage Kf tile (32 x 384 bf16 = one contiguous 24 KB block) ---
#if __has_builtin(__builtin_amdgcn_tensor_load_to_lds) && \
    __has_builtin(__builtin_amdgcn_s_wait_tensorcnt)
    if (w == 0) {                         // one wave issues the TDM descriptor
      unsigned long long gaddr =
          (unsigned long long)(size_t)(kbp + (size_t)kb * FDIM);
      unsigned int laddr = (unsigned int)(size_t)&lKf[0];
      // D# group0: count=1, lds_addr, global_addr[56:0], type=2
      tdm_u32x4 g0;
      g0.x = 1u;                                   // count=1, gather off
      g0.y = laddr;                                // lds_addr
      g0.z = (unsigned int)gaddr;                  // global_addr[31:0]
      g0.w = (unsigned int)((gaddr >> 32) & 0x1FFFFFFu) | (2u << 30); // type=2
      // D# group1: 1-D tile: data_size=2B, tensor_dim0=tile_dim0=12288,
      // tensor_dim1=tile_dim1=1, stride0=12288
      const unsigned int NE = 32 * FDIM;           // 12288 elements
      tdm_i32x8 g1;
      g1[0] = (int)(1u << 16);                     // data_size=1 (2 bytes)
      g1[1] = (int)((NE & 0xFFFFu) << 16);         // tensor_dim0 lo
      g1[2] = (int)(((NE >> 16) & 0xFFFFu) | (1u << 16)); // dim0 hi | dim1 lo
      g1[3] = (int)((NE & 0xFFFFu) << 16);         // dim1 hi=0 | tile_dim0
      g1[4] = 1;                                   // tile_dim1=1, tile_dim2=0
      g1[5] = (int)NE;                             // tensor_dim0_stride lo
      g1[6] = 0;
      g1[7] = 0;
      tdm_i32x4 gz = {0, 0, 0, 0};                 // groups 2/3 unused (<=2D)
#if defined(__clang_major__) && (__clang_major__ >= 23)
      tdm_i32x8 gz8 = {0, 0, 0, 0, 0, 0, 0, 0};
      __builtin_amdgcn_tensor_load_to_lds(g0, g1, gz, gz, gz8, 0);
#else
      __builtin_amdgcn_tensor_load_to_lds(g0, g1, gz, gz, 0);
#endif
      __builtin_amdgcn_s_wait_tensorcnt(0);        // s_wait_tensorcnt 0
    }
#else
    for (int idx = threadIdx.x; idx < 32 * (FDIM / 8); idx += 128) {
      int row = idx / (FDIM / 8), col = idx % (FDIM / 8);
      *(u32x4*)(lKf + row * FDIM + col * 8) =
          *(const u32x4*)(kbp + (size_t)(kb + row) * FDIM + col * 8);
    }
#endif
    // --- cooperative stage: V tile transposed -> lVt[d][j] ---
    for (int idx = threadIdx.x; idx < 32 * 8; idx += 128) {
      int j = idx >> 3, db = idx & 7;
      union { u32x4 u; unsigned short s[8]; } pk;
      pk.u = *(const u32x4*)(vbp + (size_t)(kb + j) * DHEAD + db * 8);
      for (int e = 0; e < 8; ++e) lVt[(db * 8 + e) * 32 + j] = pk.s[e];
    }
    __syncthreads();

    if (kb <= qbase + 15) {             // wave-uniform: EXEC all-ones for WMMA
      v8f s0 = {}, s1 = {};
      for (int kk = 0; kk < FDIM / 32; ++kk) {
        v16bf b0 = load_frag(lKf + ln * FDIM + kk * 32, hl);
        v16bf b1 = load_frag(lKf + (16 + ln) * FDIM + kk * 32, hl);
        s0 = wmma_bf16(qfr[kk], b0, s0);
        s1 = wmma_bf16(qfr[kk], b1, s1);
      }
      // causal mask (j > i -> 0) + bf16 round-trip to LDS in row-major [16][32]
      unsigned short* scw = lSc[w];
      for (int v = 0; v < 8; ++v) {
        int m = v + 8 * hl;
        int i = qbase + m;
        float x0 = (kb + ln <= i) ? s0[v] : 0.f;
        float x1 = (kb + 16 + ln <= i) ? s1[v] : 0.f;
        scw[m * 32 + ln]      = f2bf(x0);
        scw[m * 32 + 16 + ln] = f2bf(x1);
      }
      // denominator: per-row sum of the 32 masked scores
      if (lane < 16) {
        float sden = 0.f;
        for (int j = 0; j < 32; ++j) sden += bf2f(scw[lane * 32 + j]);
        lDen[w][lane] += sden;
      }
      // numerator: Sc(16x32) @ V(32x64), B-frags = rows of transposed V
      v16bf fa2 = load_frag(scw + ln * 32, hl);
      n0 = wmma_bf16(fa2, load_frag(lVt + (0 * 16 + ln) * 32, hl), n0);
      n1 = wmma_bf16(fa2, load_frag(lVt + (1 * 16 + ln) * 32, hl), n1);
      n2 = wmma_bf16(fa2, load_frag(lVt + (2 * 16 + ln) * 32, hl), n2);
      n3 = wmma_bf16(fa2, load_frag(lVt + (3 * 16 + ln) * 32, hl), n3);
    }
  }

  // write attn_vec rows: av[(s*BATCH + b)*1024 + head*64 + d]
  for (int v = 0; v < 8; ++v) {
    int m = v + 8 * hl;
    int s = qbase + m;
    float inv = 1.f / (lDen[w][m] + EPSD);
    float* orow = av + ((size_t)s * BATCH + b) * DMODEL + head * DHEAD;
    orow[0 * 16 + ln] = n0[v] * inv;
    orow[1 * 16 + ln] = n1[v] * inv;
    orow[2 * 16 + ln] = n2[v] * inv;
    orow[3 * 16 + ln] = n3[v] * inv;
  }
}

// ---------------------------------------------------------------------------
// Kernel 6: f32 -> bf16 cast (attn_vec for the output projection)
// ---------------------------------------------------------------------------
__global__ void castbf_kernel(const float* __restrict__ src,
                              unsigned short* __restrict__ dst, int n) {
  for (int i = blockIdx.x * blockDim.x + threadIdx.x; i < n;
       i += gridDim.x * blockDim.x)
    dst[i] = f2bf(src[i]);
}

// ---------------------------------------------------------------------------
// Kernel 8: residual + LayerNorm.  One block per token row (1024 cols).
// ---------------------------------------------------------------------------
__global__ __launch_bounds__(256) void ln_kernel(
    const float* __restrict__ h, const float* __restrict__ ao,
    const float* __restrict__ g, const float* __restrict__ bt,
    float* __restrict__ out) {
  __shared__ float red[256];
  int r = blockIdx.x, t = threadIdx.x;
  float x[4];
  float s = 0.f;
  for (int i = 0; i < 4; ++i) {
    int c = t + i * 256;
    x[i] = h[(size_t)r * DMODEL + c] + ao[(size_t)r * DMODEL + c];
    s += x[i];
  }
  red[t] = s; __syncthreads();
  for (int o = 128; o > 0; o >>= 1) {
    if (t < o) red[t] += red[t + o];
    __syncthreads();
  }
  float mean = red[0] * (1.f / DMODEL);
  __syncthreads();
  float vs = 0.f;
  for (int i = 0; i < 4; ++i) { float d = x[i] - mean; vs += d * d; }
  red[t] = vs; __syncthreads();
  for (int o = 128; o > 0; o >>= 1) {
    if (t < o) red[t] += red[t + o];
    __syncthreads();
  }
  float inv = rsqrtf(red[0] * (1.f / DMODEL) + EPSLN);
  for (int i = 0; i < 4; ++i) {
    int c = t + i * 256;
    out[(size_t)r * DMODEL + c] = g[c] * (x[i] - mean) * inv + bt[c];
  }
}

// ---------------------------------------------------------------------------
// Launch orchestration
// ---------------------------------------------------------------------------
extern "C" void kernel_launch(void* const* d_in, const int* in_sizes, int n_in,
                              void* d_out, int out_size, void* d_ws, size_t ws_size,
                              hipStream_t stream) {
  const float* h     = (const float*)d_in[0];
  const float* Wq    = (const float*)d_in[1];
  const float* Wkv   = (const float*)d_in[2];
  const float* Wo    = (const float*)d_in[3];
  const float* gamma = (const float*)d_in[4];
  const float* beta  = (const float*)d_in[5];
  // d_in[6] = attn_mask (causal triu) -- applied analytically in attn_kernel.
  float* out = (float*)d_out;

  char* ws = (char*)d_ws;
  size_t off = 0;
  auto alloc = [&](size_t bytes) -> void* {
    void* p = ws + off;
    off += (bytes + 255) & ~(size_t)255;
    return p;
  };
  unsigned short* hbf   = (unsigned short*)alloc((size_t)MROWS * DMODEL * 2);
  unsigned short* wqkv  = (unsigned short*)alloc((size_t)QKVCOLS * DMODEL * 2);
  unsigned short* wo    = (unsigned short*)alloc((size_t)DMODEL * DMODEL * 2);
  float*          qkv   = (float*)alloc((size_t)MROWS * QKVCOLS * 4);
  unsigned short* qf    = (unsigned short*)alloc((size_t)BATCH * NHEAD * SEQ * FDIM * 2);
  unsigned short* kf    = (unsigned short*)alloc((size_t)BATCH * NHEAD * SEQ * FDIM * 2);
  unsigned short* vbn   = (unsigned short*)alloc((size_t)BATCH * NHEAD * SEQ * DHEAD * 2);
  float*          av    = (float*)alloc((size_t)MROWS * DMODEL * 4);
  unsigned short* avbf  = (unsigned short*)alloc((size_t)MROWS * DMODEL * 2);
  float*          aout  = (float*)alloc((size_t)MROWS * DMODEL * 4);

  // 1. bf16 casts
  cast_in_kernel<<<2048, 256, 0, stream>>>(h, Wq, Wkv, Wo, hbf, wqkv, wo);

  // 2. QKV projection: [3072,1024] @ [3072,1024]^T -> [3072,3072]
  {
    int waves = (MROWS / 16) * (QKVCOLS / 64);        // 9216
    gemm_bf16_nt<<<waves / 8, 256, 0, stream>>>(hbf, wqkv, qkv,
                                                MROWS, QKVCOLS, DMODEL);
  }

  // 3. DPFP feature maps (SCALE folded into Q)
  dpfp_kernel<<<MROWS * NHEAD, 128, 0, stream>>>(qkv, 0,      qf, SCALEF);
  dpfp_kernel<<<MROWS * NHEAD, 128, 0, stream>>>(qkv, DMODEL, kf, 1.0f);

  // 4. V cast to per-(b,head) layout
  vcast_kernel<<<2048, 256, 0, stream>>>(qkv, vbn);

  // 5. streaming kernelized attention
  attn_kernel<<<BATCH * NHEAD * 24, 128, 0, stream>>>(qf, kf, vbn, av);

  // 6. cast attention vectors to bf16
  castbf_kernel<<<2048, 256, 0, stream>>>(av, avbf, MROWS * DMODEL);

  // 7. output projection: [3072,1024] @ Wo^T -> [3072,1024]
  {
    int waves = (MROWS / 16) * (DMODEL / 64);         // 3072
    gemm_bf16_nt<<<waves / 8, 256, 0, stream>>>(avbf, wo, aout,
                                                MROWS, DMODEL, DMODEL);
  }

  // 8. residual + post-LayerNorm
  ln_kernel<<<MROWS, 256, 0, stream>>>(h, aout, gamma, beta, out);
}
